// SpMiddleFHDNobn_33998961115508
// MI455X (gfx1250) — matrix operand order repaired and checked by reading, not compiled, not run on hardware
//
#include <hip/hip_runtime.h>
#include <hip/hip_bf16.h>
#include <stdint.h>

// ---------------------------------------------------------------------------
// CDNA5 (gfx1250) implicit-GEMM sparse-CNN backbone, fully layer-specialized.
// v_wmma_f32_16x16x32_bf16 with MT=2 x NT=cout/16 register tiling, weights
// staged into LDS once per workgroup via global_load_async_to_lds_b128.
// ---------------------------------------------------------------------------

typedef __attribute__((ext_vector_type(16))) __bf16 v16bf;
typedef __attribute__((ext_vector_type(8)))  float  v8f;

#define CONV_TPB 256   // 8 wave32 per workgroup

__device__ __forceinline__ uint16_t f2bf(float f) {
  uint32_t u = __float_as_uint(f);
  uint32_t r = u + 0x7FFFu + ((u >> 16) & 1u);   // round-to-nearest-even
  return (uint16_t)(r >> 16);
}

// ---- scatter voxel features into dense bf16 channels-last volume + mask ----
__global__ __launch_bounds__(256) void scatter_kernel(
    const float* __restrict__ vf, const int* __restrict__ coors,
    uint16_t* __restrict__ x0, uint8_t* __restrict__ m0,
    int NV, int H, int W) {
  int tid = blockIdx.x * blockDim.x + threadIdx.x;
  if (tid >= NV * 16) return;
  int v = tid >> 4, c = tid & 15;
  int z = coors[v * 4 + 1], y = coors[v * 4 + 2], x = coors[v * 4 + 3];
  int vox = (z * H + y) * W + x;
  x0[vox * 16 + c] = f2bf(vf[tid]);
  if (c == 0) m0[vox] = 1;
}

// ---- pre-pack weights fp32 -> pair-interleaved bf16 ------------------------
// element (k,n) stored at (k>>1)*(cout*2) + n*2 + (k&1): a (k,k+1) pair for
// column n is one aligned dword.
__global__ __launch_bounds__(256) void prep_kernel(
    const float* __restrict__ w, uint16_t* __restrict__ wb,
    int Ktot, int Kpad, int cout) {
  int i = blockIdx.x * blockDim.x + threadIdx.x;
  if (i >= Kpad * cout) return;
  int k = i / cout, n = i % cout;
  uint16_t val = (k < Ktot) ? f2bf(w[k * cout + n]) : (uint16_t)0;
  wb[(k >> 1) * (cout * 2) + n * 2 + (k & 1)] = val;
}

// ---- occupancy-mask propagation for strided (down) layers ------------------
__global__ __launch_bounds__(256) void mask_down_kernel(
    const uint8_t* __restrict__ mi, uint8_t* __restrict__ mo,
    int inD, int inH, int inW, int outD, int outH, int outW,
    int kz, int ky, int kx, int sz, int sy, int sx, int pz, int py, int px) {
  int vox = blockIdx.x * blockDim.x + threadIdx.x;
  int nv = outD * outH * outW;
  if (vox >= nv) return;
  int ox = vox % outW;
  int t  = vox / outW;
  int oy = t % outH;
  int oz = t / outH;
  uint8_t m = 0;
  for (int dz = 0; dz < kz; ++dz) {
    int iz = oz * sz + dz - pz;
    if ((unsigned)iz >= (unsigned)inD) continue;
    for (int dy = 0; dy < ky; ++dy) {
      int iy = oy * sy + dy - py;
      if ((unsigned)iy >= (unsigned)inH) continue;
      for (int dx = 0; dx < kx; ++dx) {
        int ix = ox * sx + dx - px;
        if ((unsigned)ix >= (unsigned)inW) continue;
        m |= mi[(iz * inH + iy) * inW + ix];
      }
    }
  }
  mo[vox] = m ? 1 : 0;
}

// ---- layer-specialized implicit-GEMM conv3d --------------------------------
// Workgroup: 8 wave32; weights staged once to LDS by async copy, each wave
// then computes an MT(=2)x16-voxel by COUT-channel tile.
// A fragment (16x32 bf16, ISA 7.12.2): lane holds row M=lane&15,
//   VGPR j<4 : K = hi*8 + 2j + h ; VGPR j>=4 : K = 16 + hi*8 + 2(j-4) + h.
// B fragment (32x16 bf16): lane holds column N=lane&15, K same as A fragment.
// D (16x16 f32): reg r holds row M = r + hi*8, column N = lane&15.
template <int CIN, int COUT, int KZ, int KY, int KX,
          int SZ, int SY, int SX, int PZ, int PY, int PX,
          int IND, int INH, int INW, int OUTD, int OUTH, int OUTW, bool FINAL>
__global__ __launch_bounds__(CONV_TPB) void conv_wmma(
    const uint16_t* __restrict__ in, const uint16_t* __restrict__ wb,
    const float* __restrict__ bias, const uint8_t* __restrict__ mask,
    uint16_t* __restrict__ outb, float* __restrict__ outf) {
  constexpr int NTAPS = KZ * KY * KX;
  constexpr int KTOT  = NTAPS * CIN;
  constexpr int KPAD  = (KTOT + 31) & ~31;
  constexpr int NT    = COUT / 16;
  constexpr int MT    = 2;
  constexpr int NVOX  = OUTD * OUTH * OUTW;
  constexpr int CINH  = CIN / 2;
  constexpr int WBYTES = KPAD * COUT * 2;   // packed bf16 weight block
  constexpr int WVEC   = WBYTES / 16;       // 16-byte async-copy chunks

  extern __shared__ uint16_t sW[];          // dynamic LDS: WBYTES

  // ---- stage the whole weight block to LDS (async, ASYNCcnt-tracked) ------
  {
    const char* gw = (const char*)wb;
#pragma unroll 1
    for (int i = threadIdx.x; i < WVEC; i += CONV_TPB) {
      uint32_t ldsAddr = (uint32_t)(uintptr_t)((char*)sW + i * 16);
      uint64_t gaddr   = (uint64_t)(uintptr_t)(gw + i * 16);
      asm volatile("global_load_async_to_lds_b128 %0, %1, off"
                   :: "v"(ldsAddr), "v"(gaddr) : "memory");
    }
    asm volatile("s_wait_asynccnt 0x0" ::: "memory");
    __syncthreads();
  }

  const int lane = threadIdx.x & 31;        // wave32 lane
  const int waveId = threadIdx.x >> 5;      // 0..7
  const int mrow = lane & 15;
  const int hi   = lane >> 4;
  const int tileBase = (blockIdx.x * (CONV_TPB / 32) + waveId) * (16 * MT);
  if (tileBase >= NVOX) return;             // wave-uniform exit (after barrier)

  // Decode the A-row voxel for each M-tile (constant div -> mul/shift).
  int ozA[MT], oyA[MT], oxA[MT];
  bool rowOK[MT];
#pragma unroll
  for (int mt = 0; mt < MT; ++mt) {
    int voxA = tileBase + mt * 16 + mrow;
    rowOK[mt] = voxA < NVOX;
    int v = rowOK[mt] ? voxA : 0;
    oxA[mt] = v % OUTW;
    int t = v / OUTW;
    oyA[mt] = t % OUTH;
    ozA[mt] = t / OUTH;
  }

  const uint32_t* in32 = (const uint32_t*)in;   // bf16 pairs, channels-last
  const uint32_t* sW32 = (const uint32_t*)sW;   // LDS weights, pair-interleaved

  v8f acc[MT][NT] = {};

#pragma unroll 1
  for (int kb = 0; kb < KPAD; kb += 32) {
    union { uint32_t u[8]; v16bf v; } A[MT];
#pragma unroll
    for (int j = 0; j < 8; ++j) {
      const int koff = (j < 4) ? (2 * j) : (16 + 2 * (j - 4));
      int ka = kb + hi * 8 + koff;           // pair K index (even)
      int t  = ka / CIN;                     // power-of-two -> shift
      int c  = ka % CIN;
      int dz = t / (KY * KX);                // constant divisors
      int rem = t % (KY * KX);
      int dy = rem / KX;
      int dx = rem % KX;
      bool tapOK = (t < NTAPS);              // covers KPAD tail
#pragma unroll
      for (int mt = 0; mt < MT; ++mt) {
        uint32_t av = 0;
        if (tapOK && rowOK[mt]) {
          int iz = ozA[mt] * SZ + dz - PZ;
          int iy = oyA[mt] * SY + dy - PY;
          int ix = oxA[mt] * SX + dx - PX;
          if ((unsigned)iz < (unsigned)IND && (unsigned)iy < (unsigned)INH &&
              (unsigned)ix < (unsigned)INW) {
            int voxI = (iz * INH + iy) * INW + ix;
            av = in32[voxI * CINH + (c >> 1)];  // two consecutive cin (bf16x2)
          }
        }
        A[mt].u[j] = av;
      }
    }
#pragma unroll
    for (int nt = 0; nt < NT; ++nt) {
      union { uint32_t u[8]; v16bf v; } B;
#pragma unroll
      for (int j = 0; j < 8; ++j) {
        int kbp = kb + hi * 8 + ((j < 4) ? (2 * j) : (16 + 2 * (j - 4)));
        B.u[j] = sW32[(kbp >> 1) * COUT + nt * 16 + mrow];   // ds_load
      }
#pragma unroll
      for (int mt = 0; mt < MT; ++mt) {
        acc[mt][nt] = __builtin_amdgcn_wmma_f32_16x16x32_bf16(
            false, A[mt].v, false, B.v, (short)0, acc[mt][nt], false, false);
      }
    }
  }

  // Epilogue: bias + ReLU + occupancy mask, bf16 store (or fp32 final).
#pragma unroll
  for (int nt = 0; nt < NT; ++nt) {
    const int nCol = nt * 16 + mrow;
    const float bn = bias[nCol];
#pragma unroll
    for (int mt = 0; mt < MT; ++mt) {
#pragma unroll
      for (int r = 0; r < 8; ++r) {
        int voxO = tileBase + mt * 16 + r + hi * 8;
        if (voxO < NVOX) {
          float v = acc[mt][nt][r] + bn;
          v = v > 0.f ? v : 0.f;
          v *= (float)mask[voxO];
          if (!FINAL) {
            outb[voxO * COUT + nCol] = f2bf(v);
          } else {
            // final layer: fp32 output, (C*D, H, W) contiguous order
            int ox2 = voxO % OUTW;
            int t = voxO / OUTW;
            int oy2 = t % OUTH;
            int oz2 = t / OUTH;
            outf[((nCol * OUTD + oz2) * OUTH + oy2) * OUTW + ox2] = v;
          }
        }
      }
    }
  }
}

template <int CIN, int COUT, int KZ, int KY, int KX,
          int SZ, int SY, int SX, int PZ, int PY, int PX,
          int IND, int INH, int INW, int OUTD, int OUTH, int OUTW, bool FINAL>
static void runConv(const uint16_t* in, const uint16_t* wb, const float* bias,
                    const uint8_t* mask, uint16_t* outb, float* outf,
                    hipStream_t s) {
  constexpr int NVOX = OUTD * OUTH * OUTW;
  constexpr int KPAD = (KZ * KY * KX * CIN + 31) & ~31;
  constexpr int WBYTES = KPAD * COUT * 2;
  constexpr int VOX_PER_BLOCK = (CONV_TPB / 32) * 32;   // 8 waves x 32 voxels
  dim3 grid((NVOX + VOX_PER_BLOCK - 1) / VOX_PER_BLOCK);
  conv_wmma<CIN, COUT, KZ, KY, KX, SZ, SY, SX, PZ, PY, PX,
            IND, INH, INW, OUTD, OUTH, OUTW, FINAL>
      <<<grid, CONV_TPB, WBYTES, s>>>(in, wb, bias, mask, outb, outf);
}

// ---------------------------------------------------------------------------
// Host launcher
// ---------------------------------------------------------------------------
extern "C" void kernel_launch(void* const* d_in, const int* in_sizes, int n_in,
                              void* d_out, int out_size, void* d_ws, size_t ws_size,
                              hipStream_t stream) {
  (void)n_in; (void)out_size; (void)ws_size;

  const float* vf    = (const float*)d_in[0];
  const int*   coors = (const int*)d_in[1];
  const int NV = in_sizes[0] / 16;

  const float* Wt[14];
  const float* Bs[14];
  for (int i = 0; i < 14; ++i) {
    Wt[i] = (const float*)d_in[4 + i];
    Bs[i] = (const float*)d_in[18 + i];
  }

  // Resolutions after each downsample stage.
  static const int rD[5] = {41, 21, 11, 5, 2};
  static const int rH[5] = {160, 80, 40, 20, 20};
  static const int rW[5] = {176, 88, 44, 22, 22};
  int nvoxR[5];
  for (int i = 0; i < 5; ++i) nvoxR[i] = rD[i] * rH[i] * rW[i];

  static const int cinL[14]  = {16,16,16,32,32,32,64,64,64,64,64,64,64,64};
  static const int coutL[14] = {16,16,32,32,32,64,64,64,64,64,64,64,64,64};
  static const int ktapL[14] = {27,27,27,27,27,27,27,27,27,27,27,27,27,3};

  // ---- workspace carve-up --------------------------------------------------
  char* ws = (char*)d_ws;
  size_t off = 0;
  auto alloc = [&](size_t bytes) -> void* {
    void* p = ws + off;
    off += (bytes + 255) & ~size_t(255);
    return p;
  };
  const size_t maxActBytes = (size_t)nvoxR[0] * 16 * sizeof(uint16_t);
  uint16_t* actA = (uint16_t*)alloc(maxActBytes);
  uint16_t* actB = (uint16_t*)alloc(maxActBytes);
  uint8_t* maskR[5];
  for (int i = 0; i < 5; ++i) maskR[i] = (uint8_t*)alloc(nvoxR[i]);
  uint16_t* wbL[14];
  int KtotL[14], KpadL[14];
  for (int l = 0; l < 14; ++l) {
    KtotL[l] = ktapL[l] * cinL[l];
    KpadL[l] = (KtotL[l] + 31) & ~31;
    wbL[l] = (uint16_t*)alloc((size_t)KpadL[l] * coutL[l] * sizeof(uint16_t));
  }

  // ---- layer-0 input: zero + scatter --------------------------------------
  hipMemsetAsync(actA, 0, maxActBytes, stream);
  hipMemsetAsync(maskR[0], 0, (size_t)nvoxR[0], stream);
  {
    int n = NV * 16;
    scatter_kernel<<<(n + 255) / 256, 256, 0, stream>>>(
        vf, coors, actA, maskR[0], NV, rH[0], rW[0]);
  }

  // ---- weight prep ---------------------------------------------------------
  for (int l = 0; l < 14; ++l) {
    int n = KpadL[l] * coutL[l];
    prep_kernel<<<(n + 255) / 256, 256, 0, stream>>>(
        Wt[l], wbL[l], KtotL[l], KpadL[l], coutL[l]);
  }

  // ---- mask pyramid (down layers: 2, 5, 9, 13) ----------------------------
  mask_down_kernel<<<(nvoxR[1] + 255) / 256, 256, 0, stream>>>(
      maskR[0], maskR[1], rD[0], rH[0], rW[0], rD[1], rH[1], rW[1],
      3, 3, 3, 2, 2, 2, 1, 1, 1);
  mask_down_kernel<<<(nvoxR[2] + 255) / 256, 256, 0, stream>>>(
      maskR[1], maskR[2], rD[1], rH[1], rW[1], rD[2], rH[2], rW[2],
      3, 3, 3, 2, 2, 2, 1, 1, 1);
  mask_down_kernel<<<(nvoxR[3] + 255) / 256, 256, 0, stream>>>(
      maskR[2], maskR[3], rD[2], rH[2], rW[2], rD[3], rH[3], rW[3],
      3, 3, 3, 2, 2, 2, 0, 1, 1);
  mask_down_kernel<<<(nvoxR[4] + 255) / 256, 256, 0, stream>>>(
      maskR[3], maskR[4], rD[3], rH[3], rW[3], rD[4], rH[4], rW[4],
      3, 1, 1, 2, 1, 1, 0, 0, 0);

  // ---- the 14 conv layers (ping-pong actA/actB) ---------------------------
  uint16_t* cur = actA;
  uint16_t* nxt = actB;
#define SWAP_BUFS() { uint16_t* t_ = cur; cur = nxt; nxt = t_; }

  // L0, L1: subm 16->16 @ r0
  runConv<16,16,3,3,3,1,1,1,1,1,1, 41,160,176, 41,160,176,false>(
      cur, wbL[0], Bs[0], maskR[0], nxt, nullptr, stream); SWAP_BUFS();
  runConv<16,16,3,3,3,1,1,1,1,1,1, 41,160,176, 41,160,176,false>(
      cur, wbL[1], Bs[1], maskR[0], nxt, nullptr, stream); SWAP_BUFS();
  // L2: down 16->32, r0 -> r1
  runConv<16,32,3,3,3,2,2,2,1,1,1, 41,160,176, 21,80,88,false>(
      cur, wbL[2], Bs[2], maskR[1], nxt, nullptr, stream); SWAP_BUFS();
  // L3, L4: subm 32->32 @ r1
  runConv<32,32,3,3,3,1,1,1,1,1,1, 21,80,88, 21,80,88,false>(
      cur, wbL[3], Bs[3], maskR[1], nxt, nullptr, stream); SWAP_BUFS();
  runConv<32,32,3,3,3,1,1,1,1,1,1, 21,80,88, 21,80,88,false>(
      cur, wbL[4], Bs[4], maskR[1], nxt, nullptr, stream); SWAP_BUFS();
  // L5: down 32->64, r1 -> r2
  runConv<32,64,3,3,3,2,2,2,1,1,1, 21,80,88, 11,40,44,false>(
      cur, wbL[5], Bs[5], maskR[2], nxt, nullptr, stream); SWAP_BUFS();
  // L6-L8: subm 64->64 @ r2
  runConv<64,64,3,3,3,1,1,1,1,1,1, 11,40,44, 11,40,44,false>(
      cur, wbL[6], Bs[6], maskR[2], nxt, nullptr, stream); SWAP_BUFS();
  runConv<64,64,3,3,3,1,1,1,1,1,1, 11,40,44, 11,40,44,false>(
      cur, wbL[7], Bs[7], maskR[2], nxt, nullptr, stream); SWAP_BUFS();
  runConv<64,64,3,3,3,1,1,1,1,1,1, 11,40,44, 11,40,44,false>(
      cur, wbL[8], Bs[8], maskR[2], nxt, nullptr, stream); SWAP_BUFS();
  // L9: down 64->64, r2 -> r3, pad (0,1,1)
  runConv<64,64,3,3,3,2,2,2,0,1,1, 11,40,44, 5,20,22,false>(
      cur, wbL[9], Bs[9], maskR[3], nxt, nullptr, stream); SWAP_BUFS();
  // L10-L12: subm 64->64 @ r3
  runConv<64,64,3,3,3,1,1,1,1,1,1, 5,20,22, 5,20,22,false>(
      cur, wbL[10], Bs[10], maskR[3], nxt, nullptr, stream); SWAP_BUFS();
  runConv<64,64,3,3,3,1,1,1,1,1,1, 5,20,22, 5,20,22,false>(
      cur, wbL[11], Bs[11], maskR[3], nxt, nullptr, stream); SWAP_BUFS();
  runConv<64,64,3,3,3,1,1,1,1,1,1, 5,20,22, 5,20,22,false>(
      cur, wbL[12], Bs[12], maskR[3], nxt, nullptr, stream); SWAP_BUFS();
  // L13: down 64->64, k=(3,1,1), s=(2,1,1), p=0, r3 -> r4, fp32 final output
  runConv<64,64,3,1,1,2,1,1,0,0,0, 5,20,22, 2,20,22,true>(
      cur, wbL[13], Bs[13], maskR[4], nullptr, (float*)d_out, stream);
#undef SWAP_BUFS
}
